// CrossAttentionLayer_16509854286462
// MI455X (gfx1250) — matrix operand order repaired and checked by reading, not compiled, not run on hardware
//
#include <hip/hip_runtime.h>
#include <hip/hip_bf16.h>

// ---------------- problem constants ----------------
#define EMBED 1024
#define NHEAD 16
#define HDIM  64
#define BATCH 2
#define QLEN  2048
#define CLEN  2048

typedef __bf16 bf16_t;
typedef __attribute__((ext_vector_type(16))) __bf16 v16bf;
typedef __attribute__((ext_vector_type(8)))  __bf16 v8bf;
typedef __attribute__((ext_vector_type(4)))  __bf16 v4bf;
typedef __attribute__((ext_vector_type(8)))  float  v8f;
typedef __attribute__((ext_vector_type(4)))  float  v4f;
typedef __attribute__((ext_vector_type(4)))  int    v4i;

// gfx1250 async global->LDS copy (ASYNCcnt-tracked), with safe fallback.
// Builtin signature (from hipcc diagnostic): (v4i AS1*, v4i AS3*, imm, imm).
#if defined(__AMDGCN__) && __has_builtin(__builtin_amdgcn_global_load_async_to_lds_b128)
#define HAS_ASYNC_LDS 1
__device__ __forceinline__ void async_g2l_b128(const void* g, void* l) {
  __builtin_amdgcn_global_load_async_to_lds_b128(
      (__attribute__((address_space(1))) v4i*)(g),
      (__attribute__((address_space(3))) v4i*)(l), 0, 0);
}
#define ASYNC_WAIT() asm volatile("s_wait_asynccnt 0x0" ::: "memory")
#else
#define HAS_ASYNC_LDS 0
#define ASYNC_WAIT()
#endif

__device__ __forceinline__ bf16_t f2bf(float f) {
  unsigned u = __builtin_bit_cast(unsigned, f);
  unsigned r = u + 0x7FFFu + ((u >> 16) & 1u);   // round-to-nearest-even
  unsigned short h = (unsigned short)(r >> 16);
  return __builtin_bit_cast(bf16_t, h);
}

__device__ __forceinline__ v16bf cat8(v8bf lo, v8bf hi) {
  return __builtin_shufflevector(lo, hi, 0,1,2,3,4,5,6,7,8,9,10,11,12,13,14,15);
}

// =====================================================================
// GEMM: Y[M=4096, N=1024] = A[M,1024] * W[1024,1024] + bias
//  A_BF16:    A is bf16 vs fp32 (convert while staging)
//  HEADSPLIT: store bf16 into [b][h][s][d] layout vs fp32 row-major
// Block 256 thr (8 waves). Tile 128x128, BK=32, ping-pong LDS, one
// barrier per K-step. Each wave: 16 rows x 128 cols -> 8 wmma accums.
// =====================================================================
template<bool A_BF16, bool HEADSPLIT>
__global__ __launch_bounds__(256)
void gemm_wmma(const void* __restrict__ Ap, const float* __restrict__ W,
               const float* __restrict__ bias, void* __restrict__ Out)
{
  constexpr int Kdim = EMBED, Ndim = EMBED;
  constexpr int BM = 128, BN = 128, BK = 32;
  constexpr int NKB = Kdim / BK;
  __shared__ __align__(32) bf16_t As[2][BM][40];  // row 80B
  __shared__ __align__(32) bf16_t Bs[2][BN][48];  // [n][k], row 96B

  const int tid  = threadIdx.x;
  const int wave = tid >> 5, lane = tid & 31;
  const int l16  = lane & 15, h8 = lane >> 4;
  const int mBase = blockIdx.y * BM;
  const int nBase = blockIdx.x * BN;

  // staging maps
  const int ar  = tid >> 3;          // A row base   0..31
  const int ac  = (tid & 7) * 4;     // A col        0..28
  const int bk0 = tid >> 4;          // B k base     0..15
  const int bn0 = (tid & 15) * 4;    // B n base     0..60

  auto loadA = [&](int kb, v4bf aR[4]) {
    #pragma unroll
    for (int i = 0; i < 4; ++i) {
      const int r = ar + 32 * i;
      if (A_BF16) {
        const bf16_t* A = (const bf16_t*)Ap;
        aR[i] = *(const v4bf*)(A + (size_t)(mBase + r) * Kdim + kb + ac);
      } else {
        const float* A = (const float*)Ap;
        v4f v = *(const v4f*)(A + (size_t)(mBase + r) * Kdim + kb + ac);
        v4bf o;
        o[0] = f2bf(v[0]); o[1] = f2bf(v[1]);
        o[2] = f2bf(v[2]); o[3] = f2bf(v[3]);
        aR[i] = o;
      }
    }
  };
  auto loadB = [&](int kb, v4f bR[4]) {
    #pragma unroll
    for (int i = 0; i < 2; ++i)
      #pragma unroll
      for (int j = 0; j < 2; ++j)
        bR[i * 2 + j] = *(const v4f*)(W + (size_t)(kb + bk0 + 16 * i) * Ndim
                                        + nBase + bn0 + 64 * j);
  };
  auto storeTiles = [&](int p, v4bf aR[4], v4f bR[4]) {
    #pragma unroll
    for (int i = 0; i < 4; ++i)
      *(v4bf*)&As[p][ar + 32 * i][ac] = aR[i];
    #pragma unroll
    for (int i = 0; i < 2; ++i)
      #pragma unroll
      for (int j = 0; j < 2; ++j) {
        v4f v = bR[i * 2 + j];
        const int k = bk0 + 16 * i, n = bn0 + 64 * j;
        Bs[p][n + 0][k] = f2bf(v[0]);
        Bs[p][n + 1][k] = f2bf(v[1]);
        Bs[p][n + 2][k] = f2bf(v[2]);
        Bs[p][n + 3][k] = f2bf(v[3]);
      }
  };

  v8f acc[8] = {};
  v4bf aR[4]; v4f bR[4];
  loadA(0, aR); loadB(0, bR);
  storeTiles(0, aR, bR);
  __syncthreads();

  int p = 0;
  for (int i = 0; i < NKB; ++i) {
    if (i + 1 < NKB) { loadA((i + 1) * BK, aR); loadB((i + 1) * BK, bR); }

    const int am = wave * 16 + l16;
    v8bf lo = *(const v8bf*)&As[p][am][h8 * 16];
    v8bf hi = *(const v8bf*)&As[p][am][h8 * 16 + 8];
    v16bf a = cat8(lo, hi);
    #pragma unroll
    for (int nt = 0; nt < 8; ++nt) {
      v16bf b = *(const v16bf*)&Bs[p][nt * 16 + l16][h8 * 16];
      acc[nt] = __builtin_amdgcn_wmma_f32_16x16x32_bf16(
          false, a, false, b, (short)0, acc[nt], false, false);
    }

    if (i + 1 < NKB) storeTiles(p ^ 1, aR, bR);
    __syncthreads();
    p ^= 1;
  }

  // ---- epilogue: bias + store ----
  #pragma unroll
  for (int nt = 0; nt < 8; ++nt) {
    const int n  = nBase + nt * 16 + l16;
    const float bv = bias[n];
    #pragma unroll
    for (int r = 0; r < 8; ++r) {
      const int m = mBase + wave * 16 + r + h8 * 8;   // C/D layout row
      const float val = acc[nt][r] + bv;
      if (HEADSPLIT) {
        const int bidx = m / QLEN, s = m % QLEN;
        const int h = n / HDIM, d = n % HDIM;
        bf16_t* O = (bf16_t*)Out;
        O[(((size_t)bidx * NHEAD + h) * QLEN + s) * HDIM + d] = f2bf(val);
      } else {
        float* O = (float*)Out;
        O[(size_t)m * Ndim + n] = val;
      }
    }
  }
}

// =====================================================================
// Flash attention per (128 q-rows, head, batch). Block 256 thr (8 waves),
// each wave owns 16 q-rows (Q fragments resident). Double-buffered K/V
// staging: K via async global->LDS copy, V via register-transposed store.
// Per 64-key chunk: 8 wmma (S) + 8 wmma (O += P V), online softmax.
// =====================================================================
__global__ __launch_bounds__(256)
void attn_wmma(const bf16_t* __restrict__ Qb, const bf16_t* __restrict__ Kb,
               const bf16_t* __restrict__ Vb, bf16_t* __restrict__ Att)
{
  constexpr int QT = 128;
  constexpr int KT = 64;
  constexpr int NCH = CLEN / KT;
  __shared__ __align__(32) bf16_t Ks[2][KT][80];     // [key][d]
  __shared__ __align__(32) bf16_t Vt[2][HDIM][80];   // [d][key]
  __shared__ __align__(32) bf16_t Ps[8][16][80];     // per-wave P

  const int tid  = threadIdx.x;
  const int wave = tid >> 5, lane = tid & 31;
  const int l16  = lane & 15, h8 = lane >> 4;
  const int qBase = blockIdx.x * QT;
  const int h = blockIdx.y, b = blockIdx.z;

  const size_t headQ = ((size_t)b * NHEAD + h) * (size_t)QLEN * HDIM;
  const size_t headK = ((size_t)b * NHEAD + h) * (size_t)CLEN * HDIM;

  // staging maps
  const int srow = tid >> 2;   // 0..63
  const int q4   = tid & 3;    // 16-wide quarter

  auto stageK = [&](int p, int kc) {
    const bf16_t* src = Kb + headK + (size_t)(kc + srow) * HDIM + q4 * 16;
#if HAS_ASYNC_LDS
    async_g2l_b128(src,     &Ks[p][srow][q4 * 16]);
    async_g2l_b128(src + 8, &Ks[p][srow][q4 * 16 + 8]);
#else
    *(v8bf*)&Ks[p][srow][q4 * 16]     = *(const v8bf*)(src);
    *(v8bf*)&Ks[p][srow][q4 * 16 + 8] = *(const v8bf*)(src + 8);
#endif
  };
  auto loadV = [&](int kc, v8bf vr[2]) {
    const bf16_t* src = Vb + headK + (size_t)(kc + srow) * HDIM + q4 * 16;
    vr[0] = *(const v8bf*)(src);
    vr[1] = *(const v8bf*)(src + 8);
  };
  auto storeV = [&](int p, v8bf vr[2]) {
    #pragma unroll
    for (int j = 0; j < 2; ++j)
      #pragma unroll
      for (int i = 0; i < 8; ++i)
        Vt[p][q4 * 16 + j * 8 + i][srow] = vr[j][i];
  };

  // persistent Q A-fragments (16 rows x 64 d = 2 k-chunks)
  const bf16_t* qrow = Qb + headQ + (size_t)(qBase + wave * 16 + l16) * HDIM;
  v16bf qa[2];
  #pragma unroll
  for (int c = 0; c < 2; ++c) {
    v8bf lo = *(const v8bf*)(qrow + c * 32 + h8 * 16);
    v8bf hi = *(const v8bf*)(qrow + c * 32 + h8 * 16 + 8);
    qa[c] = cat8(lo, hi);
  }

  v8f o[4] = {};
  float mrow[8], lrow[8];
  #pragma unroll
  for (int r = 0; r < 8; ++r) { mrow[r] = -1e30f; lrow[r] = 0.f; }

  // prologue: stage chunk 0
  {
    stageK(0, 0);
    v8bf vr[2];
    loadV(0, vr);
    storeV(0, vr);
    ASYNC_WAIT();
    __syncthreads();
  }

  int p = 0;
  for (int i = 0; i < NCH; ++i) {
    v8bf vr[2];
    if (i + 1 < NCH) { stageK(p ^ 1, (i + 1) * KT); loadV((i + 1) * KT, vr); }

    // ---- S = Q K^T ----
    v8f s[4] = {};
    #pragma unroll
    for (int nt = 0; nt < 4; ++nt) {
      #pragma unroll
      for (int c = 0; c < 2; ++c) {
        v16bf bk = *(const v16bf*)&Ks[p][nt * 16 + l16][c * 32 + h8 * 16];
        s[nt] = __builtin_amdgcn_wmma_f32_16x16x32_bf16(
            false, qa[c], false, bk, (short)0, s[nt], false, false);
      }
    }

    // ---- online softmax (rows live in 16-lane halves of C/D layout) ----
    #pragma unroll
    for (int r = 0; r < 8; ++r) {
      float s0 = s[0][r] * 0.125f;   // 1/sqrt(64)
      float s1 = s[1][r] * 0.125f;
      float s2 = s[2][r] * 0.125f;
      float s3 = s[3][r] * 0.125f;
      float rmax = fmaxf(fmaxf(s0, s1), fmaxf(s2, s3));
      rmax = fmaxf(rmax, __shfl_xor(rmax, 1, 32));
      rmax = fmaxf(rmax, __shfl_xor(rmax, 2, 32));
      rmax = fmaxf(rmax, __shfl_xor(rmax, 4, 32));
      rmax = fmaxf(rmax, __shfl_xor(rmax, 8, 32));
      const float mnew  = fmaxf(mrow[r], rmax);
      const float alpha = __expf(mrow[r] - mnew);
      const float p0 = __expf(s0 - mnew);
      const float p1 = __expf(s1 - mnew);
      const float p2 = __expf(s2 - mnew);
      const float p3 = __expf(s3 - mnew);
      float ls = p0 + p1 + p2 + p3;
      ls += __shfl_xor(ls, 1, 32);
      ls += __shfl_xor(ls, 2, 32);
      ls += __shfl_xor(ls, 4, 32);
      ls += __shfl_xor(ls, 8, 32);
      lrow[r] = lrow[r] * alpha + ls;
      mrow[r] = mnew;
      #pragma unroll
      for (int dt = 0; dt < 4; ++dt) o[dt][r] *= alpha;
      const int rl = r + h8 * 8;            // actual local row
      Ps[wave][rl][ 0 + l16] = f2bf(p0);
      Ps[wave][rl][16 + l16] = f2bf(p1);
      Ps[wave][rl][32 + l16] = f2bf(p2);
      Ps[wave][rl][48 + l16] = f2bf(p3);
    }
    asm volatile("s_wait_dscnt 0x0" ::: "memory");  // same-wave P write->read

    // ---- O += P V ----
    v16bf pa[2];
    #pragma unroll
    for (int c = 0; c < 2; ++c) {
      v8bf lo = *(const v8bf*)&Ps[wave][l16][c * 32 + h8 * 16];
      v8bf hi = *(const v8bf*)&Ps[wave][l16][c * 32 + h8 * 16 + 8];
      pa[c] = cat8(lo, hi);
    }
    #pragma unroll
    for (int dt = 0; dt < 4; ++dt) {
      #pragma unroll
      for (int c = 0; c < 2; ++c) {
        v16bf vb = *(const v16bf*)&Vt[p][dt * 16 + l16][c * 32 + h8 * 16];
        o[dt] = __builtin_amdgcn_wmma_f32_16x16x32_bf16(
            false, pa[c], false, vb, (short)0, o[dt], false, false);
      }
    }

    if (i + 1 < NCH) storeV(p ^ 1, vr);
    ASYNC_WAIT();
    __syncthreads();
    p ^= 1;
  }

  // ---- normalize + store attended (row-major [b][s][h*64+d], bf16) ----
  #pragma unroll
  for (int r = 0; r < 8; ++r) {
    const float inv = lrow[r] > 0.f ? 1.0f / lrow[r] : 0.f;
    const int sr = qBase + wave * 16 + r + h8 * 8;
    #pragma unroll
    for (int dt = 0; dt < 4; ++dt) {
      const int col = h * HDIM + dt * 16 + l16;
      Att[((size_t)b * QLEN + sr) * EMBED + col] = f2bf(o[dt][r] * inv);
    }
  }
}

// =====================================================================
extern "C" void kernel_launch(void* const* d_in, const int* in_sizes, int n_in,
                              void* d_out, int out_size, void* d_ws, size_t ws_size,
                              hipStream_t stream) {
  (void)in_sizes; (void)n_in; (void)out_size; (void)ws_size;
  const float* query   = (const float*)d_in[0];
  const float* context = (const float*)d_in[1];
  const float* Wq = (const float*)d_in[2];
  const float* bq = (const float*)d_in[3];
  const float* Wk = (const float*)d_in[4];
  const float* bk = (const float*)d_in[5];
  const float* Wv = (const float*)d_in[6];
  const float* bv = (const float*)d_in[7];
  const float* Wo = (const float*)d_in[8];
  const float* bo = (const float*)d_in[9];
  float* out = (float*)d_out;

  char* ws = (char*)d_ws;
  const size_t szHead = (size_t)BATCH * NHEAD * QLEN * HDIM * sizeof(bf16_t); // 8 MiB
  bf16_t* Qb = (bf16_t*)(ws);
  bf16_t* Kb = (bf16_t*)(ws + szHead);
  bf16_t* Vb = (bf16_t*)(ws + 2 * szHead);
  bf16_t* At = (bf16_t*)(ws + 3 * szHead);

  dim3 gGemm(EMBED / 128, (BATCH * QLEN) / 128);
  // projections (fp32 in -> bf16 head-split out)
  gemm_wmma<false, true><<<gGemm, 256, 0, stream>>>(query,   Wq, bq, Qb);
  gemm_wmma<false, true><<<gGemm, 256, 0, stream>>>(context, Wk, bk, Kb);
  gemm_wmma<false, true><<<gGemm, 256, 0, stream>>>(context, Wv, bv, Vb);
  // attention
  dim3 gAtt(QLEN / 128, NHEAD, BATCH);
  attn_wmma<<<gAtt, 256, 0, stream>>>(Qb, Kb, Vb, At);
  // output projection (bf16 in -> fp32 out + bias)
  gemm_wmma<true, false><<<gGemm, 256, 0, stream>>>(At, Wo, bo, out);
}
